// LSTMModel_2826088481019
// MI455X (gfx1250) — matrix operand order repaired and checked by reading, not compiled, not run on hardware
//
#include <hip/hip_runtime.h>

// ---------------------------------------------------------------------------
// 2-layer LSTM (B=2048, T=512, I=3, H=128, O=2) for MI455X / gfx1250.
// Persistent batch-tile workgroups, bf16 WMMA (v_wmma_f32_16x16x32_bf16),
// cell state in VGPRs, h double-buffered in LDS, ALL weights register-resident
// (w_hh for both layers, w_ih1 for layer 1), x_t staged through LDS.
// Activations use v_exp_f32 + v_rcp_f32 (TRANS ops co-execute with WMMA).
// ---------------------------------------------------------------------------

typedef __bf16        v16bf __attribute__((ext_vector_type(16)));
typedef float         v8f   __attribute__((ext_vector_type(8)));
typedef unsigned int  v4u   __attribute__((ext_vector_type(4)));

union Frag {
  v16bf bf;
  v4u   u4[2];
};

__device__ __forceinline__ float sigmoidf_fast(float x) {
  // 1/(1+e^-x) with hardware rcp (no IEEE div expansion)
  return __builtin_amdgcn_rcpf(1.0f + __expf(-x));
}
__device__ __forceinline__ float tanhf_fast(float x) {
  // tanh(x) = 2/(1+e^-2x) - 1
  const float r = __builtin_amdgcn_rcpf(1.0f + __expf(-2.0f * x));
  return fmaf(2.0f, r, -1.0f);
}

#define HP 144  // LDS row pitch (bf16): 288B rows -> 16B aligned, banks spread

__global__ __launch_bounds__(256) void convert_weights(
    const float* __restrict__ whh0, const float* __restrict__ wih1,
    const float* __restrict__ whh1,
    const float* __restrict__ bih0, const float* __restrict__ bhh0,
    const float* __restrict__ bih1, const float* __restrict__ bhh1,
    __bf16* __restrict__ whh0b, __bf16* __restrict__ wih1b,
    __bf16* __restrict__ whh1b,
    float* __restrict__ bias0, float* __restrict__ bias1) {
  int i = blockIdx.x * blockDim.x + threadIdx.x;
  if (i < 512 * 128) {
    whh0b[i] = (__bf16)whh0[i];
    wih1b[i] = (__bf16)wih1[i];
    whh1b[i] = (__bf16)whh1[i];
  }
  if (i < 512) {
    bias0[i] = bih0[i] + bhh0[i];
    bias1[i] = bih1[i] + bhh1[i];
  }
}

template <int LAYER>
__global__ __launch_bounds__(256) void lstm_layer(
    const float*  __restrict__ x,      // L0: (B,T,3) f32
    const __bf16* __restrict__ h_in,   // L1: (B,T,128) bf16
    const float*  __restrict__ w_ih0,  // L0: (512,3) f32
    const __bf16* __restrict__ Wrec,   // (512,128) bf16  (w_hh row-major)
    const __bf16* __restrict__ Wih,    // L1: (512,128) bf16 (w_ih1 row-major)
    const float*  __restrict__ bias,   // (512) f32  (b_ih + b_hh)
    __bf16*       __restrict__ h_out,  // L0: (B,T,128) bf16
    const float*  __restrict__ fc_w,   // L1: (2,128)
    const float*  __restrict__ fc_b,   // L1: (2)
    float*        __restrict__ out)    // L1: (B,2)
{
  constexpr int T  = 512;
  constexpr int Hh = 128;
  __shared__ __align__(16) __bf16 hbuf[2][16][HP];   // double-buffered h state
  __shared__ __align__(16) float  xstage[2][16][4];  // L0: staged x_t rows

  const int tid  = threadIdx.x;
  const int wave = tid >> 5;
  const int lane = tid & 31;
  const int nc   = lane & 15;       // column within 16-wide tile
  const int hi   = lane >> 4;       // lane half
  const int j    = wave * 16;       // this wave's hidden-column block
  const int m0   = blockIdx.x * 16; // batch tile base

  // zero h state buffer 0 (t = -1) and stage x_0
  for (int idx = tid; idx < 16 * HP; idx += 256)
    (&hbuf[0][0][0])[idx] = (__bf16)0.0f;
  if (LAYER == 0 && tid < 48) {
    const int row = tid / 3, d = tid - row * 3;
    xstage[0][row][d] =
        __builtin_nontemporal_load(x + (size_t)(m0 + row) * T * 3 + d);
  }

  const int khalfA = hi * 8;   // A-fragment K sub-offset (16-bit A layout)
  const int khalfB = hi * 8;   // B-fragment K sub-offset (same table order as A)
  const int mr     = nc;       // A row carried by this lane

  // Preload weights as B fragments: B[k][n] = W[n][k].  Register-stationary.
  Frag  Brec[4][4];
  Frag  Bih[4][4];   // layer 1 only
  float bias_r[4];
  float w3[4][3];
#pragma unroll
  for (int g = 0; g < 4; ++g) {
    const int n = g * Hh + j + nc;
    bias_r[g] = bias[n];
#pragma unroll
    for (int ks = 0; ks < 4; ++ks) {
      const __bf16* p = Wrec + (size_t)n * Hh + ks * 32 + khalfB;
      Brec[g][ks].u4[0] = *(const v4u*)(p);
      Brec[g][ks].u4[1] = *(const v4u*)(p + 16);
      if (LAYER == 1) {
        const __bf16* q = Wih + (size_t)n * Hh + ks * 32 + khalfB;
        Bih[g][ks].u4[0] = *(const v4u*)(q);
        Bih[g][ks].u4[1] = *(const v4u*)(q + 16);
      }
    }
    if (LAYER == 0) {
#pragma unroll
      for (int d = 0; d < 3; ++d) w3[g][d] = w_ih0[n * 3 + d];
    }
  }

  float c[8];
#pragma unroll
  for (int v = 0; v < 8; ++v) c[v] = 0.0f;

  __syncthreads();

  int buf = 0;
  for (int t = 0; t < T; ++t) {
    v8f acc[4];
    if (LAYER == 0) {
      // C init = bias + x_t @ w_ih0^T  (I=3), x_t rows from LDS stage
#pragma unroll
      for (int v = 0; v < 8; ++v) {
        const int m = v + hi * 8;
        const float4 xr = *(const float4*)&xstage[buf][m][0];
#pragma unroll
        for (int g = 0; g < 4; ++g)
          acc[g][v] = fmaf(xr.x, w3[g][0],
                      fmaf(xr.y, w3[g][1],
                      fmaf(xr.z, w3[g][2], bias_r[g])));
      }
    } else {
#pragma unroll
      for (int g = 0; g < 4; ++g)
#pragma unroll
        for (int v = 0; v < 8; ++v) acc[g][v] = bias_r[g];
    }

    if (LAYER == 1) {
      // fused input projection: gates += h1[:,t,:] @ w_ih1^T (K=128, global A,
      // register-resident B)
      const __bf16* hrow = h_in + ((size_t)(m0 + mr) * T + t) * Hh;
      __builtin_prefetch(hrow + Hh, 0, 0);  // next timestep row
#pragma unroll
      for (int ks = 0; ks < 4; ++ks) {
        Frag a;
        a.u4[0] = *(const v4u*)(hrow + ks * 32 + khalfA);
        a.u4[1] = *(const v4u*)(hrow + ks * 32 + khalfA + 16);
#pragma unroll
        for (int g = 0; g < 4; ++g) {
          acc[g] = __builtin_amdgcn_wmma_f32_16x16x32_bf16(
              false, a.bf, false, Bih[g][ks].bf, (short)0, acc[g], false, false);
        }
      }
    }

    // recurrent projection: gates += h_{t-1} @ w_hh^T  (K=128, A from LDS)
#pragma unroll
    for (int ks = 0; ks < 4; ++ks) {
      Frag a;
      const __bf16* hrow = &hbuf[buf][mr][0];
      a.u4[0] = *(const v4u*)(hrow + ks * 32 + khalfA);
      a.u4[1] = *(const v4u*)(hrow + ks * 32 + khalfA + 16);
#pragma unroll
      for (int g = 0; g < 4; ++g) {
        acc[g] = __builtin_amdgcn_wmma_f32_16x16x32_bf16(
            false, a.bf, false, Brec[g][ks].bf, (short)0, acc[g], false, false);
      }
    }

    // LSTM cell update; this wave owns hidden columns j..j+15, c stays in VGPRs
    const int nh = j + nc;
#pragma unroll
    for (int v = 0; v < 8; ++v) {
      const int m = v + hi * 8;
      const float ig = sigmoidf_fast(acc[0][v]);
      const float fg = sigmoidf_fast(acc[1][v]);
      const float gg = tanhf_fast(acc[2][v]);
      const float og = sigmoidf_fast(acc[3][v]);
      const float cn = fmaf(fg, c[v], ig * gg);
      c[v] = cn;
      const float  hn = og * tanhf_fast(cn);
      const __bf16 hb = (__bf16)hn;
      hbuf[buf ^ 1][m][nh] = hb;
      if (LAYER == 0)
        h_out[((size_t)(m0 + m) * T + t) * Hh + nh] = hb;
    }

    // stage x_{t+1} (layer 0): 48 threads, nontemporal (x is streamed once)
    if (LAYER == 0 && tid < 48 && (t + 1) < T) {
      const int row = tid / 3, d = tid - row * 3;
      xstage[buf ^ 1][row][d] = __builtin_nontemporal_load(
          x + ((size_t)(m0 + row) * T + (t + 1)) * 3 + d);
    }

    __syncthreads();
    buf ^= 1;
  }

  if (LAYER == 1) {
    // epilogue FC: out[m,:] = h_T[m,:] @ fc_w^T + fc_b   (O = 2, tiny)
    if (tid < 32) {
      const int m = tid >> 1;
      const int o = tid & 1;
      float s = fc_b[o];
      for (int k = 0; k < Hh; ++k)
        s += (float)hbuf[buf][m][k] * fc_w[o * Hh + k];
      out[(size_t)(m0 + m) * 2 + o] = s;
    }
  }
}

extern "C" void kernel_launch(void* const* d_in, const int* in_sizes, int n_in,
                              void* d_out, int out_size, void* d_ws, size_t ws_size,
                              hipStream_t stream) {
  const float* x     = (const float*)d_in[0];
  const float* w_ih0 = (const float*)d_in[1];
  const float* w_hh0 = (const float*)d_in[2];
  const float* b_ih0 = (const float*)d_in[3];
  const float* b_hh0 = (const float*)d_in[4];
  const float* w_ih1 = (const float*)d_in[5];
  const float* w_hh1 = (const float*)d_in[6];
  const float* b_ih1 = (const float*)d_in[7];
  const float* b_hh1 = (const float*)d_in[8];
  const float* fc_w  = (const float*)d_in[9];
  const float* fc_b  = (const float*)d_in[10];
  float* out = (float*)d_out;

  char* ws = (char*)d_ws;
  // workspace layout (bytes), all 16B-aligned
  __bf16* whh0b = (__bf16*)(ws + 0);        // 512*128 bf16 = 128KB
  __bf16* wih1b = (__bf16*)(ws + 131072);   // 128KB
  __bf16* whh1b = (__bf16*)(ws + 262144);   // 128KB
  float*  bias0 = (float*)(ws + 393216);    // 2KB
  float*  bias1 = (float*)(ws + 395264);    // 2KB
  __bf16* h1    = (__bf16*)(ws + 397312);   // B*T*H bf16 = 256MB

  convert_weights<<<256, 256, 0, stream>>>(w_hh0, w_ih1, w_hh1, b_ih0, b_hh0,
                                           b_ih1, b_hh1, whh0b, wih1b, whh1b,
                                           bias0, bias1);

  // 2048/16 = 128 persistent batch-tile workgroups, 8 waves each
  lstm_layer<0><<<128, 256, 0, stream>>>(x, nullptr, w_ih0, whh0b, nullptr,
                                         bias0, h1, nullptr, nullptr, nullptr);
  lstm_layer<1><<<128, 256, 0, stream>>>(nullptr, h1, nullptr, whh1b, wih1b,
                                         bias1, nullptr, fc_w, fc_b, out);
}
